// RPMIL_28149215658072
// MI455X (gfx1250) — compile-verified, matrix operand order checked by reading, not run on hardware
//
#include <hip/hip_runtime.h>
#include <hip/hip_bf16.h>
#include <math.h>

typedef __attribute__((ext_vector_type(16))) __bf16 v16bf;
typedef __attribute__((ext_vector_type(8)))  float  v8f;

__device__ inline __bf16 f2bf(float f) {
    union { float f; unsigned u; } v; v.f = f;
    unsigned r = (v.u + 0x7FFFu + ((v.u >> 16) & 1u)) >> 16;
    unsigned short s = (unsigned short)r;
    return __builtin_bit_cast(__bf16, s);
}

// Load a 16x32 bf16 fragment (A from row-major [16,ld] tile, or B from the
// transposed-B array whose rows are output columns). Layout per ISA 7.12.2:
// lane l -> row (l&15); half h=(l>>4): elems 0..7 = K (k0+8h .. k0+8h+7),
// elems 8..15 = K (k0+16+8h .. k0+16+8h+7).
__device__ inline v16bf load_frag(const __bf16* base, int ld, int lane, int k0) {
    int r = lane & 15, h = lane >> 4;
    const __bf16* p = base + (long)r * ld + k0 + h * 8;
    union { uint4 u; __bf16 b[8]; } c0, c1;
    c0.u = *(const uint4*)(p);
    c1.u = *(const uint4*)(p + 16);
    v16bf f;
#pragma unroll
    for (int i = 0; i < 8; ++i) { f[i] = c0.b[i]; f[8 + i] = c1.b[i]; }
    return f;
}

__device__ inline v8f wmma_bf(v16bf a, v16bf b, v8f c) {
    return __builtin_amdgcn_wmma_f32_16x16x32_bf16(false, a, false, b, (short)0, c, false, false);
}

// CDNA5 async global->LDS copy (16 bytes per lane), tracked by ASYNCcnt.
__device__ inline void async_copy16(unsigned lds_addr, const void* gaddr) {
    asm volatile("global_load_async_to_lds_b128 %0, %1, off"
                 :: "v"(lds_addr), "v"((unsigned long long)(size_t)gaddr)
                 : "memory");
}
__device__ inline void wait_async0() {
    asm volatile("s_wait_asynccnt 0" ::: "memory");
}

// ---------------- prep: row stats + bf16 X, LN(X), LN(X)^T --------------
__global__ void rp_prep_x(const float* __restrict__ X, const float* __restrict__ g,
                          const float* __restrict__ b, __bf16* Xbf, __bf16* Xnbf,
                          __bf16* Xnt) {
    int wid = threadIdx.x >> 5, lane = threadIdx.x & 31;
    int row = blockIdx.x * 8 + wid;
    const float* xr = X + (long)row * 1024;
    float s = 0.f, s2 = 0.f;
    for (int j = 0; j < 32; ++j) { float v = xr[lane + j * 32]; s += v; s2 += v * v; }
    for (int m = 1; m < 32; m <<= 1) { s += __shfl_xor(s, m); s2 += __shfl_xor(s2, m); }
    float mean = s * (1.f / 1024.f);
    float var  = s2 * (1.f / 1024.f) - mean * mean;
    float rstd = rsqrtf(var + 1e-5f);
    for (int j = 0; j < 32; ++j) {
        int d = lane + j * 32;
        float v = xr[d];
        Xbf[(long)row * 1024 + d] = f2bf(v);
        __bf16 xn = f2bf((v - mean) * rstd * g[d] + b[d]);
        Xnbf[(long)row * 1024 + d] = xn;
        Xnt[(long)d * 8192 + row] = xn;
    }
}

// ---------------- weight converts ---------------------------------------
__global__ void rp_cvt_copy(const float* __restrict__ src, __bf16* dst, int n) {
    int i = blockIdx.x * 256 + threadIdx.x;
    if (i < n) dst[i] = f2bf(src[i]);
}
__global__ void rp_cvt_trans(const float* __restrict__ src, __bf16* dst, int R, int C) {
    int i = blockIdx.x * 256 + threadIdx.x;
    if (i < R * C) { int r = i / C, c = i % C; dst[(long)c * R + r] = f2bf(src[i]); }
}

// ---------------- gated attention scores a[N] ---------------------------
// Block = 64 rows; A tile (64x1024 bf16 = 128KB) staged in LDS via async
// copies; each wave computes a 64x16 slab per column tile (8 WMMAs / k-step,
// 2KB global B traffic / k-step -> 64 FLOP per global byte).
__global__ void rp_gate(const __bf16* __restrict__ Xbf, const __bf16* __restrict__ avt,
                        const __bf16* __restrict__ aut, const float* __restrict__ av_b,
                        const float* __restrict__ au_b, const float* __restrict__ aw_w,
                        const float* __restrict__ aw_b, float* a_out) {
    extern __shared__ __bf16 Atile[];           // 64*1024 bf16
    __shared__ float rows[64];
    int tid = threadIdx.x;
    int wid = tid >> 5, lane = tid & 31;
    int mbase = blockIdx.x * 64;

    // stage A: 8192 x 16-byte chunks, async global->LDS
    unsigned lbase = (unsigned)(size_t)Atile;   // low 32 bits = LDS byte offset
    const char* gsrc = (const char*)(Xbf + (long)mbase * 1024);
    for (int i = tid; i < 8192; i += 256)
        async_copy16(lbase + i * 16, gsrc + (long)i * 16);
    if (tid < 64) rows[tid] = 0.f;
    wait_async0();
    __syncthreads();

    int half = lane >> 4, colin = lane & 15;
    float part[4][8];
#pragma unroll
    for (int mt = 0; mt < 4; ++mt)
#pragma unroll
        for (int v = 0; v < 8; ++v) part[mt][v] = 0.f;

    for (int j = 0; j < 8; ++j) {
        int ct = wid + j * 8;                   // 64 column tiles / 8 waves
        const __bf16* Bv = avt + (long)(ct * 16) * 1024;
        const __bf16* Bu = aut + (long)(ct * 16) * 1024;
        v8f accv[4], accu[4];
#pragma unroll
        for (int mt = 0; mt < 4; ++mt) {
            accv[mt] = (v8f){0.f,0.f,0.f,0.f,0.f,0.f,0.f,0.f};
            accu[mt] = (v8f){0.f,0.f,0.f,0.f,0.f,0.f,0.f,0.f};
        }
        for (int kk = 0; kk < 32; ++kk) {
            v16bf bv = load_frag(Bv, 1024, lane, kk * 32);
            v16bf bu = load_frag(Bu, 1024, lane, kk * 32);
#pragma unroll
            for (int mt = 0; mt < 4; ++mt) {
                v16bf af = load_frag(Atile + (long)(mt * 16) * 1024, 1024, lane, kk * 32);
                accv[mt] = wmma_bf(af, bv, accv[mt]);
                accu[mt] = wmma_bf(af, bu, accu[mt]);
            }
        }
        int col = ct * 16 + colin;
        float aw = aw_w[col], bv_ = av_b[col], bu_ = au_b[col];
#pragma unroll
        for (int mt = 0; mt < 4; ++mt)
#pragma unroll
            for (int v = 0; v < 8; ++v) {
                float gv = tanhf(accv[mt][v] + bv_);
                float gu = 1.f / (1.f + expf(-(accu[mt][v] + bu_)));
                part[mt][v] += gv * gu * aw;
            }
    }
#pragma unroll
    for (int mt = 0; mt < 4; ++mt)
#pragma unroll
        for (int v = 0; v < 8; ++v) {
            float s = part[mt][v];
            s += __shfl_xor(s, 1); s += __shfl_xor(s, 2);
            s += __shfl_xor(s, 4); s += __shfl_xor(s, 8);
            if (colin == 0) atomicAdd(&rows[mt * 16 + v + 8 * half], s);
        }
    __syncthreads();
    if (tid < 64) a_out[mbase + tid] = rows[tid] + aw_b[0];
}

// ---------------- softmax over a[8192] ----------------------------------
__global__ void rp_softmax_a(const float* __restrict__ a, float* p) {
    __shared__ float red[256];
    int tid = threadIdx.x;
    float m = -1e30f;
    for (int i = tid; i < 8192; i += 256) m = fmaxf(m, a[i]);
    red[tid] = m; __syncthreads();
    for (int s = 128; s > 0; s >>= 1) { if (tid < s) red[tid] = fmaxf(red[tid], red[tid + s]); __syncthreads(); }
    m = red[0]; __syncthreads();
    float sum = 0.f;
    for (int i = tid; i < 8192; i += 256) sum += expf(a[i] - m);
    red[tid] = sum; __syncthreads();
    for (int s = 128; s > 0; s >>= 1) { if (tid < s) red[tid] += red[tid + s]; __syncthreads(); }
    float inv = 1.f / red[0];
    for (int i = tid; i < 8192; i += 256) p[i] = expf(a[i] - m) * inv;
}

// ---------------- x = p^T X ---------------------------------------------
__global__ void rp_pool_x(const float* __restrict__ X, const float* __restrict__ p,
                          float* xvec, float* out) {
    int d = blockIdx.x * 256 + threadIdx.x;
    float acc = 0.f;
    for (int i = 0; i < 8192; ++i) acc += p[i] * X[(long)i * 1024 + d];
    xvec[d] = acc; out[d] = acc;
}

// ---------------- tiny VAE + LN(x_) -> Xq bf16 --------------------------
__global__ void rp_vae(const float* __restrict__ xvec, const float* __restrict__ s,
                       const float* fc1_w, const float* fc1_b,
                       const float* fc21_w, const float* fc21_b,
                       const float* fc22_w, const float* fc22_b,
                       const float* fc3_w, const float* fc3_b,
                       const float* fc4_w, const float* fc4_b,
                       const float* ln_g, const float* ln_b,
                       float* xr, __bf16* Xq, float* out) {
    __shared__ float xs[1024], h1[128], mu[64], lv[64], z[2048], h3[4096];
    int tid = threadIdx.x;                       // 128 threads
    for (int i = tid; i < 1024; i += 128) xs[i] = xvec[i];
    __syncthreads();
    { float acc = fc1_b[tid];
      for (int d = 0; d < 1024; ++d) acc += xs[d] * fc1_w[d * 128 + tid];
      h1[tid] = fmaxf(acc, 0.f); }
    __syncthreads();
    if (tid < 64) {
        float am = fc21_b[tid], al = fc22_b[tid];
        for (int d = 0; d < 128; ++d) { am += h1[d] * fc21_w[d * 64 + tid];
                                        al += h1[d] * fc22_w[d * 64 + tid]; }
        mu[tid] = am; lv[tid] = al;
        out[2050 + tid] = am; out[2114 + tid] = al;
    }
    __syncthreads();
    for (int i = tid; i < 2048; i += 128) {
        int j = i & 63;
        float zv = mu[j] + expf(0.5f * lv[j]) * s[i];
        z[i] = zv; out[2178 + i] = zv;
    }
    __syncthreads();
    for (int i = tid; i < 4096; i += 128) {
        int q = i >> 7, j = i & 127;
        float acc = fc3_b[j];
        for (int d = 0; d < 64; ++d) acc += z[q * 64 + d] * fc3_w[d * 128 + j];
        h3[i] = fmaxf(acc, 0.f);
    }
    __syncthreads();
    for (int i = tid; i < 32768; i += 128) {
        int q = i >> 10, c = i & 1023;
        float acc = fc4_b[c];
        for (int d = 0; d < 128; ++d) acc += h3[q * 128 + d] * fc4_w[d * 1024 + c];
        xr[i] = acc;
    }
    __syncthreads();
    for (int c = tid; c < 1024; c += 128) {
        float acc = 0.f;
        for (int q = 0; q < 32; ++q) acc += xr[q * 1024 + c];
        out[1024 + c] = acc * (1.f / 32.f);
    }
    if (tid < 32) {
        float mean = 0.f;
        for (int d = 0; d < 1024; ++d) mean += xr[tid * 1024 + d];
        mean *= (1.f / 1024.f);
        float var = 0.f;
        for (int d = 0; d < 1024; ++d) { float t = xr[tid * 1024 + d] - mean; var += t * t; }
        var *= (1.f / 1024.f);
        float rstd = rsqrtf(var + 1e-5f);
        for (int d = 0; d < 1024; ++d)
            Xq[tid * 1024 + d] = f2bf((xr[tid * 1024 + d] - mean) * rstd * ln_g[d] + ln_b[d]);
    }
}

// ---------------- generic tiled WMMA GEMM -------------------------------
// C[head][m][n] = scale * sum_k A[head][m][k] * Bt[head][n][k] (+ bias[n])
__global__ void rp_gemm(const __bf16* __restrict__ A, int lda, long aoff,
                        const __bf16* __restrict__ B, int ldb, long boff,
                        float* Cf, __bf16* Cb, int ldc, long coff,
                        int Mt, int Nt, int Kt, int ntiles,
                        float scale, const float* __restrict__ bias) {
    int wid = threadIdx.x >> 5, lane = threadIdx.x & 31;
    int t = blockIdx.x * 8 + wid;
    if (t >= ntiles) return;
    int per = Mt * Nt;
    int head = t / per, rem = t % per;
    int mt = rem / Nt, nt = rem % Nt;
    const __bf16* Ab = A + (long)head * aoff + (long)(mt * 16) * lda;
    const __bf16* Bb = B + (long)head * boff + (long)(nt * 16) * ldb;
    v8f acc = {0.f,0.f,0.f,0.f,0.f,0.f,0.f,0.f};
    for (int kk = 0; kk < Kt; ++kk) {
        v16bf af = load_frag(Ab, lda, lane, kk * 32);
        v16bf bf = load_frag(Bb, ldb, lane, kk * 32);
        acc = wmma_bf(af, bf, acc);
    }
    int half = lane >> 4, colin = lane & 15;
    int gcol = nt * 16 + colin;
    float bval = bias ? bias[gcol] : 0.f;
#pragma unroll
    for (int v = 0; v < 8; ++v) {
        float val = acc[v] * scale + bval;
        long idx = (long)head * coff + (long)(mt * 16 + v + 8 * half) * ldc + gcol;
        if (Cf) Cf[idx] = val; else Cb[idx] = f2bf(val);
    }
}

// ---------------- row softmax of S -> bf16 attention weights ------------
__global__ void rp_softmax_s(const float* __restrict__ S, __bf16* Wbf) {
    __shared__ float red[256];
    long base = (long)blockIdx.x * 8192;
    int tid = threadIdx.x;
    float m = -1e30f;
    for (int i = tid; i < 8192; i += 256) m = fmaxf(m, S[base + i]);
    red[tid] = m; __syncthreads();
    for (int s = 128; s > 0; s >>= 1) { if (tid < s) red[tid] = fmaxf(red[tid], red[tid + s]); __syncthreads(); }
    m = red[0]; __syncthreads();
    float sum = 0.f;
    for (int i = tid; i < 8192; i += 256) sum += expf(S[base + i] - m);
    red[tid] = sum; __syncthreads();
    for (int s = 128; s > 0; s >>= 1) { if (tid < s) red[tid] += red[tid + s]; __syncthreads(); }
    float inv = 1.f / red[0];
    for (int i = tid; i < 8192; i += 256) Wbf[base + i] = f2bf(expf(S[base + i] - m) * inv);
}

// ---------------- head: mean over queries + 2-way linear ----------------
__global__ void rp_final(const float* __restrict__ Hp, const float* __restrict__ head_w,
                         const float* __restrict__ head_b, float* out) {
    __shared__ float r0[256], r1[256];
    int tid = threadIdx.x;
    float y0 = 0.f, y1 = 0.f;
    for (int d = tid; d < 1024; d += 256) {
        float m = 0.f;
        for (int q = 0; q < 32; ++q) m += Hp[q * 1024 + d];
        m *= (1.f / 32.f);
        y0 += m * head_w[d * 2];
        y1 += m * head_w[d * 2 + 1];
    }
    r0[tid] = y0; r1[tid] = y1; __syncthreads();
    for (int s = 128; s > 0; s >>= 1) {
        if (tid < s) { r0[tid] += r0[tid + s]; r1[tid] += r1[tid + s]; }
        __syncthreads();
    }
    if (tid == 0) { out[2048] = r0[0] + head_b[0]; out[2049] = r1[0] + head_b[1]; }
}

// ---------------- workspace layout (bytes) ------------------------------
#define WS_XBF   0L
#define WS_XNBF  (WS_XBF  + 8192L*1024*2)
#define WS_XNT   (WS_XNBF + 8192L*1024*2)
#define WS_AVT   (WS_XNT  + 8192L*1024*2)
#define WS_AUT   (WS_AVT  + 1024L*1024*2)
#define WS_KBF   (WS_AUT  + 1024L*1024*2)
#define WS_QT    (WS_KBF  + 1024L*3072*2)
#define WS_VT    (WS_QT   + 3072L*1024*2)
#define WS_OT    (WS_VT   + 3072L*1024*2)
#define WS_A     (WS_OT   + 1024L*3072*2)
#define WS_P     (WS_A    + 8192L*4)
#define WS_XVEC  (WS_P    + 8192L*4)
#define WS_XR    (WS_XVEC + 1024L*4)
#define WS_XQ    (WS_XR   + 32L*1024*4)
#define WS_QBF   (WS_XQ   + 32L*1024*2)
#define WS_QTIL  (WS_QBF  + 32L*3072*2)
#define WS_S     (WS_QTIL + 3L*32*1024*2)
#define WS_WBF   (WS_S    + 3L*32*8192*4)
#define WS_TBF   (WS_WBF  + 3L*32*8192*2)
#define WS_HC    (WS_TBF  + 3L*32*1024*2)
#define WS_HP    (WS_HC   + 32L*3072*2)

extern "C" void kernel_launch(void* const* d_in, const int* in_sizes, int n_in,
                              void* d_out, int out_size, void* d_ws, size_t ws_size,
                              hipStream_t stream) {
    const float* X      = (const float*)d_in[0];
    const float* s      = (const float*)d_in[1];
    const float* av_w   = (const float*)d_in[2];
    const float* av_b   = (const float*)d_in[3];
    const float* au_w   = (const float*)d_in[4];
    const float* au_b   = (const float*)d_in[5];
    const float* aw_w   = (const float*)d_in[6];
    const float* aw_b   = (const float*)d_in[7];
    const float* fc1_w  = (const float*)d_in[8];
    const float* fc1_b  = (const float*)d_in[9];
    const float* fc21_w = (const float*)d_in[10];
    const float* fc21_b = (const float*)d_in[11];
    const float* fc22_w = (const float*)d_in[12];
    const float* fc22_b = (const float*)d_in[13];
    const float* fc3_w  = (const float*)d_in[14];
    const float* fc3_b  = (const float*)d_in[15];
    const float* fc4_w  = (const float*)d_in[16];
    const float* fc4_b  = (const float*)d_in[17];
    const float* ln_g   = (const float*)d_in[18];
    const float* ln_b   = (const float*)d_in[19];
    const float* q_w    = (const float*)d_in[20];
    const float* k_w    = (const float*)d_in[21];
    const float* v_w    = (const float*)d_in[22];
    const float* o_w    = (const float*)d_in[23];
    const float* o_b    = (const float*)d_in[24];
    const float* head_w = (const float*)d_in[25];
    const float* head_b = (const float*)d_in[26];
    float* out = (float*)d_out;

    char* ws = (char*)d_ws;
    __bf16* Xbf  = (__bf16*)(ws + WS_XBF);
    __bf16* Xnbf = (__bf16*)(ws + WS_XNBF);
    __bf16* Xnt  = (__bf16*)(ws + WS_XNT);
    __bf16* avt  = (__bf16*)(ws + WS_AVT);
    __bf16* aut  = (__bf16*)(ws + WS_AUT);
    __bf16* kbf  = (__bf16*)(ws + WS_KBF);
    __bf16* qt   = (__bf16*)(ws + WS_QT);
    __bf16* vt   = (__bf16*)(ws + WS_VT);
    __bf16* ot   = (__bf16*)(ws + WS_OT);
    float*  a    = (float*)(ws + WS_A);
    float*  p    = (float*)(ws + WS_P);
    float*  xvec = (float*)(ws + WS_XVEC);
    float*  xr   = (float*)(ws + WS_XR);
    __bf16* Xq   = (__bf16*)(ws + WS_XQ);
    __bf16* Qbf  = (__bf16*)(ws + WS_QBF);
    __bf16* Qtil = (__bf16*)(ws + WS_QTIL);
    float*  S    = (float*)(ws + WS_S);
    __bf16* Wbf  = (__bf16*)(ws + WS_WBF);
    __bf16* Tbf  = (__bf16*)(ws + WS_TBF);
    __bf16* Hc   = (__bf16*)(ws + WS_HC);
    float*  Hp   = (float*)(ws + WS_HP);

    // stage 0: bf16 conversions (X + LN stats, transposed weights)
    rp_prep_x<<<1024, 256, 0, stream>>>(X, ln_g, ln_b, Xbf, Xnbf, Xnt);
    rp_cvt_trans<<<(1024*1024 + 255) / 256, 256, 0, stream>>>(av_w, avt, 1024, 1024);
    rp_cvt_trans<<<(1024*1024 + 255) / 256, 256, 0, stream>>>(au_w, aut, 1024, 1024);
    rp_cvt_copy <<<(1024*3072 + 255) / 256, 256, 0, stream>>>(k_w, kbf, 1024 * 3072);
    rp_cvt_trans<<<(1024*3072 + 255) / 256, 256, 0, stream>>>(q_w, qt, 1024, 3072);
    rp_cvt_trans<<<(1024*3072 + 255) / 256, 256, 0, stream>>>(v_w, vt, 1024, 3072);
    rp_cvt_trans<<<(3072*1024 + 255) / 256, 256, 0, stream>>>(o_w, ot, 3072, 1024);

    // stage 1: gated MIL pooling (64-row LDS-staged blocks, async copies)
    rp_gate<<<128, 256, 64 * 1024 * sizeof(__bf16), stream>>>(
        Xbf, avt, aut, av_b, au_b, aw_w, aw_b, a);
    rp_softmax_a<<<1, 256, 0, stream>>>(a, p);
    rp_pool_x<<<4, 256, 0, stream>>>(X, p, xvec, out);

    // stage 2: VAE (+ writes mean(x_), mu, logvar, z outputs; LN(x_)->Xq)
    rp_vae<<<1, 128, 0, stream>>>(xvec, s, fc1_w, fc1_b, fc21_w, fc21_b,
                                  fc22_w, fc22_b, fc3_w, fc3_b, fc4_w, fc4_b,
                                  ln_g, ln_b, xr, Xq, out);

    // stage 3: attention, restructured as  S = (Q Wk^T) Xn^T ;  h = (W Xn) Wv
    // Q = Xq @ q_w        : [32,3072]
    rp_gemm<<<48, 256, 0, stream>>>(Xq, 1024, 0, qt, 1024, 0,
                                    nullptr, Qbf, 3072, 0,
                                    2, 192, 32, 384, 1.f, nullptr);
    // Qtil_h = Q_h @ Wk_h^T : 3 x [32,1024]
    rp_gemm<<<48, 256, 0, stream>>>(Qbf, 3072, 1024, kbf, 3072, 1024,
                                    nullptr, Qtil, 1024, 32 * 1024,
                                    2, 64, 32, 384, 1.f, nullptr);
    // S_h = Qtil_h @ Xn^T / 32 : 3 x [32,8192]
    rp_gemm<<<384, 256, 0, stream>>>(Qtil, 1024, 32 * 1024, Xnbf, 1024, 0,
                                     S, nullptr, 8192, 32L * 8192,
                                     2, 512, 32, 3072, 1.f / 32.f, nullptr);
    rp_softmax_s<<<96, 256, 0, stream>>>(S, Wbf);
    // T_h = W_h @ Xn : 3 x [32,1024]
    rp_gemm<<<48, 256, 0, stream>>>(Wbf, 8192, 32L * 8192, Xnt, 8192, 0,
                                    nullptr, Tbf, 1024, 32 * 1024,
                                    2, 64, 256, 384, 1.f, nullptr);
    // Hc[:, h*1024+d] = T_h @ Wv_h : [32,3072]
    rp_gemm<<<48, 256, 0, stream>>>(Tbf, 1024, 32 * 1024, vt, 1024, 1024L * 1024,
                                    nullptr, Hc, 3072, 1024,
                                    2, 64, 32, 384, 1.f, nullptr);
    // Hp = Hc @ o_w + o_b : [32,1024]
    rp_gemm<<<16, 256, 0, stream>>>(Hc, 3072, 0, ot, 3072, 0,
                                    Hp, nullptr, 1024, 0,
                                    2, 64, 96, 128, 1.f, o_b);

    // stage 4: y_hat
    rp_final<<<1, 256, 0, stream>>>(Hp, head_w, head_b, out);
}
